// SelfAttention_57724360458947
// MI455X (gfx1250) — compile-verified, matrix operand order checked by reading, not run on hardware
//
#include <hip/hip_runtime.h>
#include <hip/hip_bf16.h>

#define NB 4
#define LL 2048
#define EE 1024
#define HH 16
#define DD 64

typedef __attribute__((ext_vector_type(16))) _Float16 v16h;
typedef __attribute__((ext_vector_type(8)))  _Float16 v8h;
typedef __attribute__((ext_vector_type(8)))  float    v8f;
typedef __attribute__((ext_vector_type(4)))  int      v4i;

// 1/sqrt(E) * log2(e): softmax computed with exp2 (hardware-native) is exact
// softmax of the reference logits since all logits share the same scale.
#define SCALE_LOG2 (0.03125f * 1.44269504088896340736f)

static __device__ __forceinline__ v16h cat16(v8h lo, v8h hi) {
  v16h r;
#pragma unroll
  for (int i = 0; i < 8; ++i) { r[i] = lo[i]; r[i + 8] = hi[i]; }
  return r;
}

static __device__ __forceinline__ v8h cvt8(const float* p) {
  v8h r;
#pragma unroll
  for (int i = 0; i < 8; ++i) r[i] = (_Float16)p[i];
  return r;
}

static __device__ __forceinline__ v8f wmma16(v16h a, v16h b, v8f c) {
  // D(16x16,f32) = A(16x32,f16) * B(32x16,f16) + C
  return __builtin_amdgcn_wmma_f32_16x16x32_f16(false, a, false, b, (short)0, c,
                                                false, false);
}

// ---------------------------------------------------------------- Wo -> f16
__global__ void k_prep_wo(const float* __restrict__ Wo, _Float16* __restrict__ WoH) {
  int i = blockIdx.x * blockDim.x + threadIdx.x;
  if (i < EE * EE) WoH[i] = (_Float16)Wo[i];
}

// ------------------------------------------- per-head projections Q/K/V @ W^T
// Qp, Kp : f16 [N][H][L][D] row-major.  VpT : f16 [N][H][D][L] (head-transposed)
__global__ void k_proj(const float* __restrict__ Q, const float* __restrict__ K,
                       const float* __restrict__ V,
                       const float* __restrict__ Wq, const float* __restrict__ Wk,
                       const float* __restrict__ Wv,
                       _Float16* __restrict__ Qp, _Float16* __restrict__ Kp,
                       _Float16* __restrict__ VpT) {
  const int lane = threadIdx.x & 31;
  const int wave = threadIdx.x >> 5;
  const int tile = blockIdx.x * 8 + wave;      // 3 * 8192 tiles total
  const int mat  = tile >> 13;                  // 0=Q 1=K 2=V
  int rem        = tile & 8191;
  const int n    = rem >> 11;                   // / (H*128)
  rem           &= 2047;
  const int h    = rem >> 7;
  const int q0   = (rem & 127) * 16;
  const int col  = lane & 15;
  const int kb   = (lane >> 4) * 8;
  const int hi   = lane >> 4;

  const float* X = (mat == 0) ? Q : (mat == 1) ? K : V;
  const float* W = (mat == 0) ? Wq : (mat == 1) ? Wk : Wv;

  // A fragments: X tile (16 rows x 64 d), f32 -> f16 inline
  const float* xrow = X + ((size_t)(n * LL + q0 + col)) * EE + h * DD;
  const v16h a0 = cat16(cvt8(xrow + kb),      cvt8(xrow + 16 + kb));
  const v16h a1 = cat16(cvt8(xrow + 32 + kb), cvt8(xrow + 48 + kb));

  const size_t nh = (size_t)(n * HH + h);
#pragma unroll
  for (int j = 0; j < 4; ++j) {
    // B^T = W (row e, contiguous d)
    const float* wrow = W + (size_t)(j * 16 + col) * DD;
    const v16h b0 = cat16(cvt8(wrow + kb),      cvt8(wrow + 16 + kb));
    const v16h b1 = cat16(cvt8(wrow + 32 + kb), cvt8(wrow + 48 + kb));
    v8f c = {};
    c = wmma16(a0, b0, c);
    c = wmma16(a1, b1, c);
    if (mat == 2) {
      // transpose store: per lane the 8 C values are contiguous along L
      v8h pk;
#pragma unroll
      for (int r = 0; r < 8; ++r) pk[r] = (_Float16)c[r];
      *(v8h*)(VpT + (nh * DD + j * 16 + col) * LL + q0 + 8 * hi) = pk;
    } else {
      _Float16* base = ((mat == 0) ? Qp : Kp) + nh * LL * DD;
#pragma unroll
      for (int r = 0; r < 8; ++r)
        base[(size_t)(q0 + r + 8 * hi) * DD + j * 16 + col] = (_Float16)c[r];
    }
  }
}

// ------------------------------------------------------- flash attention core
// one wave per (n, h, 16-query tile); Ctx: f16 [N][L][E]
__global__ void k_attn(const _Float16* __restrict__ Qp, const _Float16* __restrict__ Kp,
                       const _Float16* __restrict__ VpT, const int* __restrict__ mask,
                       _Float16* __restrict__ Ctx) {
  const int lane = threadIdx.x & 31;
  const int wave = threadIdx.x >> 5;
  const int tile = blockIdx.x * 8 + wave;      // 8192 tiles
  const int n    = tile >> 11;
  int rem        = tile & 2047;
  const int h    = rem >> 7;
  const int q0   = (rem & 127) * 16;
  const int col  = lane & 15;
  const int kb   = (lane >> 4) * 8;
  const int hi   = lane >> 4;
  const size_t nh = (size_t)(n * HH + h);

  // B fragments of Q^T (== Q rows in A-pattern), fixed for the whole loop
  const _Float16* qrow = Qp + (nh * LL + q0 + col) * DD;
  const v16h bq0 = cat16(*(const v8h*)(qrow + kb),      *(const v8h*)(qrow + 16 + kb));
  const v16h bq1 = cat16(*(const v8h*)(qrow + 32 + kb), *(const v8h*)(qrow + 48 + kb));

  float m_run = -3.4e38f, l_run = 0.f;   // running max in log2 domain
  v8f o0 = {}, o1 = {}, o2 = {}, o3 = {};

  const int*      mbase = mask + (size_t)n * LL * LL + (size_t)(q0 + col) * LL + 8 * hi;
  const _Float16* kbase = Kp + nh * LL * DD;
  const _Float16* vbase = VpT + (nh * DD + col) * LL;

#pragma unroll 1
  for (int k0 = 0; k0 < LL; k0 += 32) {
    // S^T = K * Q^T : two 16x16 tiles (k 0-15, 16-31)
    const _Float16* kr0 = kbase + (size_t)(k0 + col) * DD;
    const _Float16* kr1 = kr0 + 16 * DD;
    const v16h ak00 = cat16(*(const v8h*)(kr0 + kb),      *(const v8h*)(kr0 + 16 + kb));
    const v16h ak01 = cat16(*(const v8h*)(kr0 + 32 + kb), *(const v8h*)(kr0 + 48 + kb));
    const v16h ak10 = cat16(*(const v8h*)(kr1 + kb),      *(const v8h*)(kr1 + 16 + kb));
    const v16h ak11 = cat16(*(const v8h*)(kr1 + 32 + kb), *(const v8h*)(kr1 + 48 + kb));
    v8f sT0 = {}, sT1 = {};
    sT0 = wmma16(ak00, bq0, sT0);  sT0 = wmma16(ak01, bq1, sT0);
    sT1 = wmma16(ak10, bq0, sT1);  sT1 = wmma16(ak11, bq1, sT1);

    // mask (k contiguous per lane); reference: (-1e20 if masked) then /sqrt(E);
    // whole row pre-scaled by log2(e) so exp2 gives exact reference softmax
    const int* mr = mbase + k0;
    const v4i mA = *(const v4i*)(mr);
    const v4i mB = *(const v4i*)(mr + 4);
    const v4i mC = *(const v4i*)(mr + 16);
    const v4i mD = *(const v4i*)(mr + 20);

    float s0[8], s1[8];
#pragma unroll
    for (int r = 0; r < 8; ++r) {
      const int mv0 = (r < 4) ? mA[r & 3] : mB[r & 3];
      const int mv1 = (r < 4) ? mC[r & 3] : mD[r & 3];
      s0[r] = (mv0 != 0 ? sT0[r] : -1.0e20f) * SCALE_LOG2;
      s1[r] = (mv1 != 0 ? sT1[r] : -1.0e20f) * SCALE_LOG2;
    }

    // online softmax: per-lane partial over 16 k, one xor-16 combine
    float tmax = fmaxf(s0[0], s1[0]);
#pragma unroll
    for (int r = 1; r < 8; ++r) { tmax = fmaxf(tmax, s0[r]); tmax = fmaxf(tmax, s1[r]); }
    tmax = fmaxf(tmax, __shfl_xor(tmax, 16, 32));
    const float m_new = fmaxf(m_run, tmax);
    const float corr  = exp2f(m_run - m_new);

    float tsum = 0.f;
    v16h pf;                                     // P^T B-fragment, in registers
#pragma unroll
    for (int r = 0; r < 8; ++r) {
      const float p0 = exp2f(s0[r] - m_new);
      const float p1 = exp2f(s1[r] - m_new);
      tsum += p0 + p1;
      pf[r]     = (_Float16)p0;
      pf[r + 8] = (_Float16)p1;
    }
    tsum += __shfl_xor(tsum, 16, 32);
    l_run = l_run * corr + tsum;
    m_run = m_new;
#pragma unroll
    for (int r = 0; r < 8; ++r) { o0[r] *= corr; o1[r] *= corr; o2[r] *= corr; o3[r] *= corr; }

    // O^T += V^T * P^T  (V^T A-fragments contiguous thanks to [N][H][D][L])
    const _Float16* v0 = vbase + k0;
    const v16h av0 = cat16(*(const v8h*)(v0 + kb),           *(const v8h*)(v0 + 16 + kb));
    const v16h av1 = cat16(*(const v8h*)(v0 + 16*LL + kb),   *(const v8h*)(v0 + 16*LL + 16 + kb));
    const v16h av2 = cat16(*(const v8h*)(v0 + 32*LL + kb),   *(const v8h*)(v0 + 32*LL + 16 + kb));
    const v16h av3 = cat16(*(const v8h*)(v0 + 48*LL + kb),   *(const v8h*)(v0 + 48*LL + 16 + kb));
    o0 = wmma16(av0, pf, o0);
    o1 = wmma16(av1, pf, o1);
    o2 = wmma16(av2, pf, o2);
    o3 = wmma16(av3, pf, o3);
  }

  const float rinv = 1.0f / l_run;
  _Float16* crow = Ctx + ((size_t)(n * LL) + q0 + col) * EE + h * DD + 8 * hi;
  v8h pk;
#pragma unroll
  for (int r = 0; r < 8; ++r) pk[r] = (_Float16)(o0[r] * rinv);
  *(v8h*)(crow) = pk;
#pragma unroll
  for (int r = 0; r < 8; ++r) pk[r] = (_Float16)(o1[r] * rinv);
  *(v8h*)(crow + 16) = pk;
#pragma unroll
  for (int r = 0; r < 8; ++r) pk[r] = (_Float16)(o2[r] * rinv);
  *(v8h*)(crow + 32) = pk;
#pragma unroll
  for (int r = 0; r < 8; ++r) pk[r] = (_Float16)(o3[r] * rinv);
  *(v8h*)(crow + 48) = pk;
}

// ---------------------------------------------- out = Ctx @ Wo^T + bo  (f32)
__global__ void k_out(const _Float16* __restrict__ Ctx, const _Float16* __restrict__ WoH,
                      const float* __restrict__ bo, float* __restrict__ Out) {
  const int lane = threadIdx.x & 31;
  const int wave = threadIdx.x >> 5;
  const int tile = blockIdx.x * 8 + wave;   // 512 m-tiles x 16 n-tiles
  const int m0 = (tile >> 4) * 16;
  const int e0 = (tile & 15) * 64;
  const int col = lane & 15;
  const int kb  = (lane >> 4) * 8;
  const int hi  = lane >> 4;

  v8f acc0 = {}, acc1 = {}, acc2 = {}, acc3 = {};
  const _Float16* arow = Ctx + (size_t)(m0 + col) * EE;
  const _Float16* w0   = WoH + (size_t)(e0 + col) * EE;

#pragma unroll 1
  for (int f0 = 0; f0 < EE; f0 += 32) {
    const v16h a  = cat16(*(const v8h*)(arow + f0 + kb), *(const v8h*)(arow + f0 + 16 + kb));
    const v16h b0 = cat16(*(const v8h*)(w0 + f0 + kb),            *(const v8h*)(w0 + f0 + 16 + kb));
    const v16h b1 = cat16(*(const v8h*)(w0 + 16*EE + f0 + kb),    *(const v8h*)(w0 + 16*EE + f0 + 16 + kb));
    const v16h b2 = cat16(*(const v8h*)(w0 + 32*EE + f0 + kb),    *(const v8h*)(w0 + 32*EE + f0 + 16 + kb));
    const v16h b3 = cat16(*(const v8h*)(w0 + 48*EE + f0 + kb),    *(const v8h*)(w0 + 48*EE + f0 + 16 + kb));
    acc0 = wmma16(a, b0, acc0);
    acc1 = wmma16(a, b1, acc1);
    acc2 = wmma16(a, b2, acc2);
    acc3 = wmma16(a, b3, acc3);
  }

  // straight-line epilogue: keep accumulators in VGPRs (no indexable array)
  {
    const int e = e0 + col;
    const float bias = bo[e];
#pragma unroll
    for (int r = 0; r < 8; ++r)
      Out[(size_t)(m0 + r + 8 * hi) * EE + e] = acc0[r] + bias;
  }
  {
    const int e = e0 + 16 + col;
    const float bias = bo[e];
#pragma unroll
    for (int r = 0; r < 8; ++r)
      Out[(size_t)(m0 + r + 8 * hi) * EE + e] = acc1[r] + bias;
  }
  {
    const int e = e0 + 32 + col;
    const float bias = bo[e];
#pragma unroll
    for (int r = 0; r < 8; ++r)
      Out[(size_t)(m0 + r + 8 * hi) * EE + e] = acc2[r] + bias;
  }
  {
    const int e = e0 + 48 + col;
    const float bias = bo[e];
#pragma unroll
    for (int r = 0; r < 8; ++r)
      Out[(size_t)(m0 + r + 8 * hi) * EE + e] = acc3[r] + bias;
  }
}

extern "C" void kernel_launch(void* const* d_in, const int* in_sizes, int n_in,
                              void* d_out, int out_size, void* d_ws, size_t ws_size,
                              hipStream_t stream) {
  const float* values  = (const float*)d_in[0];
  const float* keys    = (const float*)d_in[1];
  const float* queries = (const float*)d_in[2];
  const int*   mask    = (const int*)d_in[3];
  const float* Wq      = (const float*)d_in[4];
  const float* Wk      = (const float*)d_in[5];
  const float* Wv      = (const float*)d_in[6];
  const float* Wo      = (const float*)d_in[7];
  const float* bo      = (const float*)d_in[8];
  float* out = (float*)d_out;

  _Float16* ws = (_Float16*)d_ws;
  const size_t per = (size_t)NB * HH * LL * DD;   // 8M halves = 16 MB each
  _Float16* Qp  = ws;
  _Float16* Kp  = Qp + per;
  _Float16* VpT = Kp + per;
  _Float16* Ctx = VpT + per;
  _Float16* WoH = Ctx + per;                       // + 2 MB  (total ~66 MB)

  hipLaunchKernelGGL(k_prep_wo, dim3((EE * EE) / 256), dim3(256), 0, stream, Wo, WoH);
  hipLaunchKernelGGL(k_proj, dim3(3 * 8192 / 8), dim3(256), 0, stream,
                     queries, keys, values, Wq, Wk, Wv, Qp, Kp, VpT);
  hipLaunchKernelGGL(k_attn, dim3(8192 / 8), dim3(256), 0, stream,
                     Qp, Kp, VpT, mask, Ctx);
  hipLaunchKernelGGL(k_out, dim3(8192 / 8), dim3(256), 0, stream, Ctx, WoH, bo, out);
}
